// GroupedQueryAttention_24601572672039
// MI455X (gfx1250) — compile-verified, matrix operand order checked by reading, not run on hardware
//
#include <hip/hip_runtime.h>
#include <hip/hip_bf16.h>

#define SEQ    2048
#define HID    2048
#define NHEADS 32
#define NKV    8
#define HD     64
#define BATCH  2
#define MROWS  (BATCH * SEQ)   // 4096

typedef __bf16 bf16;
typedef __attribute__((ext_vector_type(8)))  bf16  v8bf;
typedef __attribute__((ext_vector_type(16))) bf16  v16bf;
typedef __attribute__((ext_vector_type(8)))  float v8f;

__device__ __forceinline__ bf16 f2bf(float f) {
  union { float f; unsigned u; } cv; cv.f = f;
  unsigned r = cv.u + 0x7fffu + ((cv.u >> 16) & 1u);   // round-to-nearest-even
  union { unsigned short s; bf16 b; } o; o.s = (unsigned short)(r >> 16);
  return o.b;
}

__device__ __forceinline__ v8f wmma_bf16(v16bf a, v16bf b, v8f c) {
  // D(f32 16x16) = A(16x32 bf16) * B(32x16 bf16) + C
  return __builtin_amdgcn_wmma_f32_16x16x32_bf16(false, a, false, b, (short)0, c,
                                                 false, false);
}

// Load one 16x32 bf16 WMMA operand fragment (A role, or B role when the
// source is stored transposed so K is contiguous). ISA 7.12.2 layout:
// lanes 0-15 hold K={0..7,16..23}, lanes 16-31 hold K={8..15,24..31};
// the M/N index is lane&15 in both halves.
__device__ __forceinline__ v16bf load_frag(const bf16* __restrict__ base, int ld,
                                           int lr, int kOff) {
  const bf16* p = base + (size_t)lr * ld + kOff;
  v8bf lo = *(const v8bf*)(p);        // global_load_b128
  v8bf hi = *(const v8bf*)(p + 16);   // global_load_b128
  v16bf r;
#pragma unroll
  for (int i = 0; i < 8; ++i) { r[i] = lo[i]; r[i + 8] = hi[i]; }
  return r;
}

__global__ void cvt_f32_bf16(const float* __restrict__ in, bf16* __restrict__ out, int n) {
  int stride = gridDim.x * blockDim.x;
  for (int i = blockIdx.x * blockDim.x + threadIdx.x; i < n; i += stride)
    out[i] = f2bf(in[i]);
}

// Y = X(MxK) * W(NxK)^T + bias.  One wave per 64x64 output tile:
// 4 A-frags x 4 B-frags -> 16 WMMAs per 32-deep K step (1:1 WMMA:b128-load,
// 4x the operand reuse of a single-tile wave).
// Yb != null: store bf16, scattered to head layout ([b,h,s,d] or [b,h,d,s]).
// Yf != null: store f32 row-major MxN (final projection -> d_out).
__launch_bounds__(256)
__global__ void gemm_wmma_bf16(const bf16* __restrict__ X, const bf16* __restrict__ W,
                               const float* __restrict__ bias,
                               bf16* __restrict__ Yb, float* __restrict__ Yf,
                               int M, int N, int K, int Hn, int transposeOut) {
  int lane = threadIdx.x & 31;
  int wave = blockIdx.x * (blockDim.x >> 5) + (threadIdx.x >> 5);
  int tilesN = N >> 6;
  if (wave >= (M >> 6) * tilesN) return;           // wave-uniform: EXEC stays all-ones
  int tm = (wave / tilesN) << 6;
  int tn = (wave % tilesN) << 6;
  int hf = lane >> 4, lr = lane & 15, kOff = hf << 3;

  const bf16* Xp = X + (size_t)tm * K;
  const bf16* Wp = W + (size_t)tn * K;
  v8f acc[4][4];
#pragma unroll
  for (int mi = 0; mi < 4; ++mi)
#pragma unroll
    for (int ni = 0; ni < 4; ++ni) acc[mi][ni] = (v8f){};

  for (int k0 = 0; k0 < K; k0 += 32) {
    __builtin_prefetch(Xp + (size_t)lr * K + k0 + 512, 0, 1);  // global_prefetch_b8
    v16bf a[4], b[4];
#pragma unroll
    for (int mi = 0; mi < 4; ++mi)
      a[mi] = load_frag(Xp + (size_t)(mi * 16) * K + k0, K, lr, kOff);
#pragma unroll
    for (int ni = 0; ni < 4; ++ni)
      b[ni] = load_frag(Wp + (size_t)(ni * 16) * K + k0, K, lr, kOff);
#pragma unroll
    for (int mi = 0; mi < 4; ++mi)
#pragma unroll
      for (int ni = 0; ni < 4; ++ni)
        acc[mi][ni] = wmma_bf16(a[mi], b[ni], acc[mi][ni]);
  }

#pragma unroll
  for (int mi = 0; mi < 4; ++mi)
#pragma unroll
    for (int ni = 0; ni < 4; ++ni)
#pragma unroll
      for (int r = 0; r < 8; ++r) {
        int row = tm + mi * 16 + r + (hf << 3);   // C/D layout: row = vgpr + 8*half
        int col = tn + ni * 16 + lr;              // col = lane&15
        float v = acc[mi][ni][r] + bias[col];
        if (Yf) {
          Yf[(size_t)row * N + col] = v;
        } else {
          int bb = row >> 11, s = row & (SEQ - 1);   // SEQ = 2048
          int h  = col >> 6,  d = col & (HD - 1);    // HD  = 64
          size_t idx = transposeOut
              ? ((size_t)(bb * Hn + h) * HD + d) * SEQ + s    // V^T: [b,hkv,d,s]
              : ((size_t)(bb * Hn + h) * SEQ + s) * HD + d;   // Q/K: [b,h,s,d]
          Yb[idx] = f2bf(v);
        }
      }
}

// Flash attention: one wave per 16-row query tile of one (b,h).
// Streams 32-key tiles: 4 score WMMAs, online softmax, LDS transpose of P,
// 4 output WMMAs against V^T fragments. Causal: only lower-triangular tiles.
__launch_bounds__(256)
__global__ void gqa_flash_wmma(const bf16* __restrict__ Qb, const bf16* __restrict__ Kb,
                               const bf16* __restrict__ Vt, bf16* __restrict__ Cb) {
  __shared__ bf16 ldsP[8][16 * 32];                // per-wave P staging, 8 KB total
  int lane  = threadIdx.x & 31;
  int wslot = threadIdx.x >> 5;
  int wave  = blockIdx.x * (blockDim.x >> 5) + wslot;
  const int qtiles = SEQ / 16;                     // 128
  int qt = wave & (qtiles - 1);
  int h  = (wave / qtiles) & (NHEADS - 1);
  int b  = wave / (qtiles * NHEADS);
  int hk = h >> 2;                                 // GROUPS = 32/8 = 4
  int q0 = qt << 4;

  const bf16* Qp = Qb + ((size_t)(b * NHEADS + h) * SEQ + q0) * HD;
  const bf16* Kp = Kb + (size_t)(b * NKV + hk) * SEQ * HD;
  const bf16* Vp = Vt + (size_t)(b * NKV + hk) * HD * SEQ;

  int hf = lane >> 4, lr = lane & 15, kOff = hf << 3;

  v16bf aq0 = load_frag(Qp, HD, lr, kOff);         // Q rows, d 0..31
  v16bf aq1 = load_frag(Qp + 32, HD, lr, kOff);    // Q rows, d 32..63

  v8f o0 = {}, o1 = {}, o2 = {}, o3 = {};
  float mrow[8], lrow[8];
#pragma unroll
  for (int r = 0; r < 8; ++r) { mrow[r] = -3.0e38f; lrow[r] = 0.f; }

  bf16* pl = &ldsP[wslot][0];
  const float LOG2E = 1.44269504088896f;
  int jEnd = ((q0 + 15) >> 5) << 5;                // last 32-key tile intersecting causal band

  for (int j = 0; j <= jEnd; j += 32) {
    // ---- scores S = (Q K^T) / 8 : two 16x16 f32 tiles ----
    v8f s0 = {}, s1 = {};
    {
      v16bf bk;
      bk = load_frag(Kp + (size_t)j * HD,             HD, lr, kOff); s0 = wmma_bf16(aq0, bk, s0);
      bk = load_frag(Kp + (size_t)j * HD + 32,        HD, lr, kOff); s0 = wmma_bf16(aq1, bk, s0);
      bk = load_frag(Kp + (size_t)(j + 16) * HD,      HD, lr, kOff); s1 = wmma_bf16(aq0, bk, s1);
      bk = load_frag(Kp + (size_t)(j + 16) * HD + 32, HD, lr, kOff); s1 = wmma_bf16(aq1, bk, s1);
    }
    bool needMask = (j + 31) > q0;
#pragma unroll
    for (int r = 0; r < 8; ++r) {
      int row = q0 + r + (hf << 3);
      float v0 = s0[r] * 0.125f;                   // 1/sqrt(64)
      float v1 = s1[r] * 0.125f;
      if (needMask) {
        if (j + lr > row)      v0 = -3.0e38f;
        if (j + 16 + lr > row) v1 = -3.0e38f;
      }
      // row max across the 16-lane half-group (rows live within one half)
      float mx = fmaxf(v0, v1);
      mx = fmaxf(mx, __shfl_xor(mx, 1, 32));
      mx = fmaxf(mx, __shfl_xor(mx, 2, 32));
      mx = fmaxf(mx, __shfl_xor(mx, 4, 32));
      mx = fmaxf(mx, __shfl_xor(mx, 8, 32));
      float mnew  = fmaxf(mrow[r], mx);
      float alpha = exp2f((mrow[r] - mnew) * LOG2E);
      mrow[r] = mnew;
      float p0 = exp2f((v0 - mnew) * LOG2E);
      float p1 = exp2f((v1 - mnew) * LOG2E);
      float rs = p0 + p1;
      rs += __shfl_xor(rs, 1, 32);
      rs += __shfl_xor(rs, 2, 32);
      rs += __shfl_xor(rs, 4, 32);
      rs += __shfl_xor(rs, 8, 32);
      lrow[r] = lrow[r] * alpha + rs;
      o0[r] *= alpha; o1[r] *= alpha; o2[r] *= alpha; o3[r] *= alpha;
      int prow = r + (hf << 3);
      pl[prow * 32 + lr]      = f2bf(p0);          // ds_store_b16
      pl[prow * 32 + 16 + lr] = f2bf(p1);
    }
    asm volatile("s_wait_dscnt 0" ::: "memory");   // same-wave DS ordering fence
    // ---- reread P in A-fragment layout (16x32 bf16) ----
    v16bf ap;
    {
      const bf16* pp = pl + lr * 32 + kOff;
      v8bf lo = *(const v8bf*)(pp);                // ds_load_b128
      v8bf hi = *(const v8bf*)(pp + 16);
#pragma unroll
      for (int i = 0; i < 8; ++i) { ap[i] = lo[i]; ap[i + 8] = hi[i]; }
    }
    // ---- O += P * V  (V^T stored [d, s]: K-contiguous B fragments) ----
    v16bf bv;
    bv = load_frag(Vp + j,            SEQ, lr, kOff); o0 = wmma_bf16(ap, bv, o0);
    bv = load_frag(Vp + 16 * SEQ + j, SEQ, lr, kOff); o1 = wmma_bf16(ap, bv, o1);
    bv = load_frag(Vp + 32 * SEQ + j, SEQ, lr, kOff); o2 = wmma_bf16(ap, bv, o2);
    bv = load_frag(Vp + 48 * SEQ + j, SEQ, lr, kOff); o3 = wmma_bf16(ap, bv, o3);
  }

  // ---- normalize and store context rows as bf16 [b,s, h*64+d] ----
#pragma unroll
  for (int r = 0; r < 8; ++r) {
    float inv = 1.0f / lrow[r];
    int row = q0 + r + (hf << 3);
    bf16* op = Cb + ((size_t)b * SEQ + row) * HID + (size_t)h * HD + lr;
    op[0]  = f2bf(o0[r] * inv);
    op[16] = f2bf(o1[r] * inv);
    op[32] = f2bf(o2[r] * inv);
    op[48] = f2bf(o3[r] * inv);
  }
}

extern "C" void kernel_launch(void* const* d_in, const int* in_sizes, int n_in,
                              void* d_out, int out_size, void* d_ws, size_t ws_size,
                              hipStream_t stream) {
  (void)in_sizes; (void)n_in; (void)out_size; (void)ws_size;
  const float* x  = (const float*)d_in[0];
  // d_in[1] is the causal mask; causality is applied analytically in-kernel.
  const float* Wq = (const float*)d_in[2];
  const float* bq = (const float*)d_in[3];
  const float* Wk = (const float*)d_in[4];
  const float* bk = (const float*)d_in[5];
  const float* Wv = (const float*)d_in[6];
  const float* bv = (const float*)d_in[7];
  const float* Wo = (const float*)d_in[8];
  const float* bo = (const float*)d_in[9];
  float* out = (float*)d_out;

  char* w = (char*)d_ws;
  size_t off = 0;
  auto alloc = [&](size_t bytes) {
    char* p = w + off; off += (bytes + 255) & ~(size_t)255; return p;
  };
  bf16* xb  = (bf16*)alloc((size_t)MROWS * HID * 2);        // 16 MB
  bf16* wqb = (bf16*)alloc((size_t)HID * HID * 2);          //  8 MB
  bf16* wkb = (bf16*)alloc((size_t)(NKV * HD) * HID * 2);   //  2 MB
  bf16* wvb = (bf16*)alloc((size_t)(NKV * HD) * HID * 2);   //  2 MB
  bf16* wob = (bf16*)alloc((size_t)HID * HID * 2);          //  8 MB
  bf16* Qb  = (bf16*)alloc((size_t)MROWS * HID * 2);        // 16 MB [b,h,s,d]
  bf16* Kb  = (bf16*)alloc((size_t)MROWS * NKV * HD * 2);   //  4 MB [b,hk,s,d]
  bf16* Vtb = (bf16*)alloc((size_t)MROWS * NKV * HD * 2);   //  4 MB [b,hk,d,s]
  bf16* Cb  = (bf16*)alloc((size_t)MROWS * HID * 2);        // 16 MB context

  // fp32 -> bf16 conversions (bandwidth-bound, negligible vs GEMMs)
  cvt_f32_bf16<<<2048, 256, 0, stream>>>(x,  xb,  MROWS * HID);
  cvt_f32_bf16<<<2048, 256, 0, stream>>>(Wq, wqb, HID * HID);
  cvt_f32_bf16<<<512,  256, 0, stream>>>(Wk, wkb, NKV * HD * HID);
  cvt_f32_bf16<<<512,  256, 0, stream>>>(Wv, wvb, NKV * HD * HID);
  cvt_f32_bf16<<<2048, 256, 0, stream>>>(Wo, wob, HID * HID);

  // projections: 64x64-per-wave tiles = (M/64)*(N/64) waves, 8 waves per block
  gemm_wmma_bf16<<<256, 256, 0, stream>>>(xb, wqb, bq, Qb,  nullptr,
                                          MROWS, HID,      HID, NHEADS, 0);
  gemm_wmma_bf16<<<64,  256, 0, stream>>>(xb, wkb, bk, Kb,  nullptr,
                                          MROWS, NKV * HD, HID, NKV,    0);
  gemm_wmma_bf16<<<64,  256, 0, stream>>>(xb, wvb, bv, Vtb, nullptr,
                                          MROWS, NKV * HD, HID, NKV,    1);

  // flash attention: B*H*(S/16) = 8192 waves -> 1024 blocks of 8 waves
  gqa_flash_wmma<<<1024, 256, 0, stream>>>(Qb, Kb, Vtb, Cb);

  // output projection, f32 result straight into d_out
  gemm_wmma_bf16<<<256, 256, 0, stream>>>(Cb, wob, bo, nullptr, out,
                                          MROWS, HID, HID, 0, 0);
}
